// GraphStateEncoder_53652731462311
// MI455X (gfx1250) — matrix operand order repaired and checked by reading, not compiled, not run on hardware
//
#include <hip/hip_runtime.h>

typedef __attribute__((ext_vector_type(16))) __bf16 v16bf;
typedef __attribute__((ext_vector_type(8)))  __bf16 v8bf;
typedef __attribute__((ext_vector_type(4)))  __bf16 v4bf;
typedef __attribute__((ext_vector_type(8)))  float  v8f;

#define N_NODES 50000
#define N_EDGES 400000
#define NODE_F 128
#define EDGE_F 64
#define HID 128

union AFrag { v8bf h[2]; v16bf v; };

__device__ __forceinline__ v8f wmma_bf16(v16bf a, v16bf b, v8f c) {
  // D(f32 16x16) = A(16x32 bf16) * B(32x16 bf16) + C  -> v_wmma_f32_16x16x32_bf16
  return __builtin_amdgcn_wmma_f32_16x16x32_bf16(
      /*neg_a=*/false, a, /*neg_b=*/false, b,
      /*c_mod=*/(short)0, c, /*reuse_a=*/false, /*reuse_b=*/false);
}

// A-fragment for 16x16x32 bf16: lane holds K = hi*8+0..7 (VGPR0-3) and
// K = 16+hi*8+0..7 (VGPR4-7) of row mn.  Two aligned 16B LDS loads.
__device__ __forceinline__ v16bf load_a(const __bf16* __restrict__ rowp,
                                        int k0, int hi) {
  AFrag a;
  a.h[0] = *(const v8bf*)(rowp + k0 + hi * 8);
  a.h[1] = *(const v8bf*)(rowp + k0 + 16 + hi * 8);
  return a.v;
}

// B-fragment: pre-swizzled so each lane's 16 bf16 are contiguous (32B).
__device__ __forceinline__ v16bf load_b(const __bf16* __restrict__ Ws,
                                        int kblk, int ntile, int lane) {
  return *(const v16bf*)(Ws + (((size_t)kblk * 8 + ntile) * 32 + lane) * 16);
}

// Convert f32 weight (K x 128, row-major) to bf16 in B-fragment order:
// dword t = (((kblk*8 + ntile)*32 + lane)*8 + v) holds rows (kb, kb+1) at
// column ntile*16 + (lane&15), kb = kblk*32 + (v>>2)*16 + (lane>>4)*8 + 2(v&3).
__global__ void swizzle_w_kernel(const float* __restrict__ W, int K,
                                 __bf16* __restrict__ SW) {
  int t = blockIdx.x * blockDim.x + threadIdx.x;
  if (t >= K * 64) return;          // K*128/2 dwords
  int v = t & 7;
  int lane = (t >> 3) & 31;
  int ntile = (t >> 8) & 7;
  int kblk = t >> 11;
  int n = ntile * 16 + (lane & 15);
  int kb = kblk * 32 + (v >> 2) * 16 + (lane >> 4) * 8 + 2 * (v & 3);
  SW[2 * t]     = (__bf16)W[(size_t)kb * HID + n];
  SW[2 * t + 1] = (__bf16)W[(size_t)(kb + 1) * HID + n];
}

// Fused 2-layer MLP: relu(X@W1 + b1) @ W2 + b2, bf16 mma / f32 accum.
// Optional f32 output (Out) and bf16 output (Obf).
// 256 threads (8 wave32), 32 rows per block. K in {64,128}.
__global__ __launch_bounds__(256) void mlp2_kernel(
    const float* __restrict__ X, int M, int K,
    const __bf16* __restrict__ W1s, const float* __restrict__ b1,
    const __bf16* __restrict__ W2s, const float* __restrict__ b2,
    float* __restrict__ Out, __bf16* __restrict__ Obf) {
  __shared__ __align__(16) __bf16 sA[32 * 128];
  __shared__ __align__(16) __bf16 sH[32 * HID];
  const int tid = threadIdx.x;
  const int m0 = blockIdx.x * 32;
  const int K4 = K >> 2;

  for (int idx = tid; idx < 32 * K4; idx += 256) {
    int r = idx / K4, c4 = idx - r * K4;
    int row = m0 + r;
    float4 x = make_float4(0.f, 0.f, 0.f, 0.f);
    if (row < M) x = ((const float4*)X)[(size_t)row * K4 + c4];
    v4bf h;
    h[0] = (__bf16)x.x; h[1] = (__bf16)x.y;
    h[2] = (__bf16)x.z; h[3] = (__bf16)x.w;
    *(v4bf*)&sA[r * K + 4 * c4] = h;
  }
  __syncthreads();

  const int wave = tid >> 5, lane = tid & 31;
  const int wv = wave & 3;   // column-tile group
  const int wg = wave >> 2;  // row-half: rows wg*16 .. wg*16+15
  const int hi = lane >> 4, mn = lane & 15;
  const __bf16* sArow = sA + (size_t)(wg * 16 + mn) * K;
  const __bf16* sHrow = sH + (size_t)(wg * 16 + mn) * HID;

  for (int t = wv; t < 8; t += 4) {
    v8f c = {};
    for (int kb = 0; kb < (K >> 5); ++kb)
      c = wmma_bf16(load_a(sArow, kb * 32, hi), load_b(W1s, kb, t, lane), c);
    const float bias = b1[t * 16 + mn];
    for (int r = 0; r < 8; ++r) {
      float v = c[r] + bias;
      sH[(size_t)(wg * 16 + r + hi * 8) * HID + t * 16 + mn] =
          (__bf16)(v > 0.f ? v : 0.f);
    }
  }
  __syncthreads();

  for (int t = wv; t < 8; t += 4) {
    v8f c = {};
    for (int kb = 0; kb < (HID >> 5); ++kb)
      c = wmma_bf16(load_a(sHrow, kb * 32, hi), load_b(W2s, kb, t, lane), c);
    const float bias = b2[t * 16 + mn];
    for (int r = 0; r < 8; ++r) {
      int row = m0 + wg * 16 + r + hi * 8;
      if (row < M) {
        float v = c[r] + bias;
        size_t o = (size_t)row * HID + t * 16 + mn;
        if (Out) Out[o] = v;
        if (Obf) Obf[o] = (__bf16)v;
      }
    }
  }
}

// Per-edge message MLP (fwd + bwd reuse one gathered LDS tile); atomic f32
// scatter-add into acc.  Embeddings arrive as bf16; the gather is a pure
// byte copy done with GLOBAL_LOAD_ASYNC_TO_LDS_B128 (ASYNCcnt-tracked).
// 256 threads, 32 edges per block (N_EDGES % 32 == 0: no guards).
__global__ __launch_bounds__(256) void message_kernel(
    const __bf16* __restrict__ node_bf, const __bf16* __restrict__ edge_bf,
    const int* __restrict__ edge_list,
    const __bf16* __restrict__ W1s, const float* __restrict__ b1,
    const __bf16* __restrict__ W2s, const float* __restrict__ b2,
    float* __restrict__ acc) {
  __shared__ __align__(16) __bf16 sA[32 * 384];   // [h_s | edge | h_d], 24KB
  __shared__ __align__(16) __bf16 sH[32 * HID];   // 8KB
  __shared__ int sIdx[64];                        // [0..31]=src, [32..63]=dst
  const int tid = threadIdx.x;
  const int e0 = blockIdx.x * 32;

  if (tid < 64) sIdx[tid] = edge_list[2 * (e0 + (tid & 31)) + (tid >> 5)];
  __syncthreads();

  // Async gather: 32 rows x 48 16B-chunks (h_s, edge, h_d) -> LDS.
  for (int idx = tid; idx < 32 * 48; idx += 256) {
    int r = idx / 48, c = idx - r * 48;
    int seg = c >> 4, cc = c & 15;
    const __bf16* g =
        (seg == 0) ? node_bf + (size_t)sIdx[r] * HID + cc * 8
      : (seg == 1) ? edge_bf + (size_t)(e0 + r) * HID + cc * 8
                   : node_bf + (size_t)sIdx[32 + r] * HID + cc * 8;
    unsigned lds = (unsigned)(size_t)&sA[r * 384 + c * 8];
    unsigned long long ga = (unsigned long long)(size_t)g;
    asm volatile("global_load_async_to_lds_b128 %0, %1, off"
                 :: "v"(lds), "v"(ga) : "memory");
  }
  asm volatile("s_wait_asynccnt 0x0" ::: "memory");

  const int wave = tid >> 5, lane = tid & 31;
  const int wv = wave & 3, wg = wave >> 2;
  const int hi = lane >> 4, mn = lane & 15;
  const __bf16* sArow = sA + (size_t)(wg * 16 + mn) * 384;
  const __bf16* sHrow = sH + (size_t)(wg * 16 + mn) * HID;

  for (int dir = 0; dir < 2; ++dir) {  // 0: [s,e,d]->dst, 1: [d,e,s]->src
    __syncthreads();
    // Stage 1: H = relu(A@W1 + b1), K=384 = 12 K-blocks of 32.
    // bwd permutes the 128-col blocks 0<->2 (4 K-blocks each).
    for (int t = wv; t < 8; t += 4) {
      v8f c = {};
      for (int kb = 0; kb < 12; ++kb) {
        int skb = dir ? ((kb < 4) ? kb + 8 : ((kb < 8) ? kb : kb - 8)) : kb;
        c = wmma_bf16(load_a(sArow, skb * 32, hi), load_b(W1s, kb, t, lane), c);
      }
      const float bias = b1[t * 16 + mn];
      for (int r = 0; r < 8; ++r) {
        float v = c[r] + bias;
        sH[(size_t)(wg * 16 + r + hi * 8) * HID + t * 16 + mn] =
            (__bf16)(v > 0.f ? v : 0.f);
      }
    }
    __syncthreads();
    // Stage 2: m = H@W2 + b2, scatter-add to target nodes.
    for (int t = wv; t < 8; t += 4) {
      v8f c = {};
      for (int kb = 0; kb < (HID >> 5); ++kb)
        c = wmma_bf16(load_a(sHrow, kb * 32, hi), load_b(W2s, kb, t, lane), c);
      const float bias = b2[t * 16 + mn];
      for (int r = 0; r < 8; ++r) {
        int rr = wg * 16 + r + hi * 8;
        int node = dir ? sIdx[rr] : sIdx[32 + rr];
        atomicAdd(&acc[(size_t)node * HID + t * 16 + mn], c[r] + bias);
      }
    }
  }
}

__global__ void copy_kernel(const float* __restrict__ src,
                            float* __restrict__ dst, int n4) {
  int i = blockIdx.x * blockDim.x + threadIdx.x;
  if (i < n4) ((float4*)dst)[i] = ((const float4*)src)[i];
}

extern "C" void kernel_launch(void* const* d_in, const int* in_sizes, int n_in,
                              void* d_out, int out_size, void* d_ws, size_t ws_size,
                              hipStream_t stream) {
  const float* node_features = (const float*)d_in[0];
  const int*   edge_list     = (const int*)d_in[1];
  const float* edge_features = (const float*)d_in[2];
  const float* ne_w1 = (const float*)d_in[4];
  const float* ne_b1 = (const float*)d_in[5];
  const float* ne_w2 = (const float*)d_in[6];
  const float* ne_b2 = (const float*)d_in[7];
  const float* ee_w1 = (const float*)d_in[8];
  const float* ee_b1 = (const float*)d_in[9];
  const float* ee_w2 = (const float*)d_in[10];
  const float* ee_b2 = (const float*)d_in[11];
  const float* ml_w1 = (const float*)d_in[12];
  const float* ml_b1 = (const float*)d_in[13];
  const float* ml_w2 = (const float*)d_in[14];
  const float* ml_b2 = (const float*)d_in[15];
  const float* agg_w1 = (const float*)d_in[16];
  const float* agg_b1 = (const float*)d_in[17];
  const float* agg_w2 = (const float*)d_in[18];
  const float* agg_b2 = (const float*)d_in[19];

  // Workspace: f32 node_emb + acc, bf16 embeddings, bf16 swizzled weights.
  float* node_emb = (float*)d_ws;                           //  25.6 MB
  float* acc      = node_emb + (size_t)N_NODES * HID;       //  25.6 MB
  __bf16* node_bf = (__bf16*)(acc + (size_t)N_NODES * HID); //  12.8 MB
  __bf16* edge_bf = node_bf + (size_t)N_NODES * HID;        // 102.4 MB
  __bf16* bw = edge_bf + (size_t)N_EDGES * HID;             //  ~440 KB
  __bf16* s_ne_w1 = bw;                 bw += NODE_F * HID;
  __bf16* s_ne_w2 = bw;                 bw += HID * HID;
  __bf16* s_ee_w1 = bw;                 bw += EDGE_F * HID;
  __bf16* s_ee_w2 = bw;                 bw += HID * HID;
  __bf16* s_ml_w1 = bw;                 bw += 2 * 3 * HID * HID;
  __bf16* s_ml_w2 = bw;                 bw += 2 * HID * HID;
  __bf16* s_agg_w1 = bw;                bw += HID * HID;
  __bf16* s_agg_w2 = bw;

  auto swz = [&](const float* W, int K, __bf16* SW) {
    int dwords = K * 64;
    swizzle_w_kernel<<<(dwords + 255) / 256, 256, 0, stream>>>(W, K, SW);
  };
  swz(ne_w1, NODE_F, s_ne_w1);
  swz(ne_w2, HID, s_ne_w2);
  swz(ee_w1, EDGE_F, s_ee_w1);
  swz(ee_w2, HID, s_ee_w2);
  for (int l = 0; l < 2; ++l) {
    swz(ml_w1 + (size_t)l * 3 * HID * HID, 3 * HID, s_ml_w1 + (size_t)l * 3 * HID * HID);
    swz(ml_w2 + (size_t)l * HID * HID, HID, s_ml_w2 + (size_t)l * HID * HID);
  }
  swz(agg_w1, HID, s_agg_w1);
  swz(agg_w2, HID, s_agg_w2);

  const dim3 blk(256);
  const int node_tiles = (N_NODES + 31) / 32;  // 1563
  const int edge_tiles = N_EDGES / 32;         // 12500

  // Encoders: node -> f32 + bf16; edges -> bf16 only.
  mlp2_kernel<<<node_tiles, blk, 0, stream>>>(
      node_features, N_NODES, NODE_F, s_ne_w1, ne_b1, s_ne_w2, ne_b2,
      node_emb, node_bf);
  mlp2_kernel<<<edge_tiles, blk, 0, stream>>>(
      edge_features, N_EDGES, EDGE_F, s_ee_w1, ee_b1, s_ee_w2, ee_b2,
      nullptr, edge_bf);

  for (int l = 0; l < 2; ++l) {
    const int n4 = N_NODES * HID / 4;
    copy_kernel<<<(n4 + 255) / 256, 256, 0, stream>>>(node_emb, acc, n4);
    message_kernel<<<edge_tiles, blk, 0, stream>>>(
        node_bf, edge_bf, edge_list,
        s_ml_w1 + (size_t)l * 3 * HID * HID, ml_b1 + l * HID,
        s_ml_w2 + (size_t)l * HID * HID,     ml_b2 + l * HID, acc);
    // Aggregation MLP; layer 0 refreshes node_emb (f32) + node_bf (bf16),
    // layer 1 writes the final f32 output.
    float*  outf = (l == 1) ? (float*)d_out : node_emb;
    __bf16* outb = (l == 1) ? nullptr : node_bf;
    mlp2_kernel<<<node_tiles, blk, 0, stream>>>(
        acc, N_NODES, HID, s_agg_w1, agg_b1, s_agg_w2, agg_b2, outf, outb);
  }
}
